// ContrastiveLoss_26087631356709
// MI455X (gfx1250) — matrix operand order repaired and checked by reading, not compiled, not run on hardware
//
#include <hip/hip_runtime.h>

// ---------------------------------------------------------------------------
// Contrastive loss over pairwise cosine similarities, fused GEMM epilogue.
// N = 8192 embeddings, D = 512.  S = E_norm @ E_norm^T computed with
// bf16 WMMA (2-term split: hi*hi + hi*lo + lo*hi, fp32 accumulate).
// ---------------------------------------------------------------------------

#define N_EMB 8192
#define D_EMB 512
#define MARGIN 0.5f
#define EPS_NRM 1e-12f

typedef __attribute__((ext_vector_type(16))) __bf16 v16bf;
typedef __attribute__((ext_vector_type(8)))  __bf16 v8bf;
typedef __attribute__((ext_vector_type(8)))  float  v8f;

__device__ __forceinline__ unsigned short f2bf_rne(float f) {
    unsigned u = __float_as_uint(f);
    unsigned r = u + 0x7fffu + ((u >> 16) & 1u);   // round-to-nearest-even
    return (unsigned short)(r >> 16);
}
__device__ __forceinline__ float bf2f(unsigned short h) {
    return __uint_as_float(((unsigned)h) << 16);
}

// --------------------------- kernel 1: normalize ---------------------------
// One block per row: e = x / max(||x||, eps); store bf16 hi + bf16 residual lo.
__global__ void cl_normalize(const float* __restrict__ x,
                             unsigned short* __restrict__ ehi,
                             unsigned short* __restrict__ elo) {
    __shared__ float red[256];
    const int row = blockIdx.x;
    const int tid = threadIdx.x;
    const float v0 = x[row * D_EMB + tid];
    const float v1 = x[row * D_EMB + 256 + tid];
    red[tid] = v0 * v0 + v1 * v1;
    __syncthreads();
    #pragma unroll
    for (int s = 128; s > 0; s >>= 1) {
        if (tid < s) red[tid] += red[tid + s];
        __syncthreads();
    }
    const float nrm = __builtin_sqrtf(red[0]);
    const float scale = 1.0f / fmaxf(nrm, EPS_NRM);

    const float e0 = v0 * scale;
    const float e1 = v1 * scale;
    const unsigned short h0 = f2bf_rne(e0);
    const unsigned short h1 = f2bf_rne(e1);
    ehi[row * D_EMB + tid]       = h0;
    ehi[row * D_EMB + 256 + tid] = h1;
    elo[row * D_EMB + tid]       = f2bf_rne(e0 - bf2f(h0));
    elo[row * D_EMB + 256 + tid] = f2bf_rne(e1 - bf2f(h1));
}

// ------------------- fragment loaders (row-major bf16, ld=512) -------------
// A 16x32 bf16 fragment (ISA 7.12.2): lane L<16 holds M=L, K={kb..kb+7, kb+16..kb+23};
// lane L>=16 holds M=L-16, K shifted by +8.  -> two contiguous 16B loads.
__device__ __forceinline__ v16bf load_a_frag(const unsigned short* __restrict__ P,
                                             int rowBase, int kb, int lane) {
    const int r  = rowBase + (lane & 15);
    const int k2 = kb + ((lane >> 4) << 3);
    const unsigned short* p = P + r * D_EMB + k2;
    v8bf x0 = *(const v8bf*)(p);
    v8bf x1 = *(const v8bf*)(p + 16);
    v16bf a;
    #pragma unroll
    for (int i = 0; i < 8; ++i) { a[i] = x0[i]; a[i + 8] = x1[i]; }
    return a;
}
// B 32x16 bf16 fragment, B = E^T: lane L<16 holds N=L (i.e. E row colBase+L),
// K={kb..kb+7, kb+16..kb+23}; lane L>=16 same row, K shifted by +8.  -> two contiguous 16B loads.
__device__ __forceinline__ v16bf load_b_frag(const unsigned short* __restrict__ P,
                                             int colBase, int kb, int lane) {
    const int r  = colBase + (lane & 15);
    const int k2 = kb + ((lane >> 4) << 3);
    const unsigned short* p = P + r * D_EMB + k2;
    v8bf x0 = *(const v8bf*)(p);
    v8bf x1 = *(const v8bf*)(p + 16);
    v16bf b;
    #pragma unroll
    for (int i = 0; i < 8; ++i) { b[i] = x0[i]; b[i + 8] = x1[i]; }
    return b;
}

#define WMMA_BF16(A, B, C)                                                    \
    __builtin_amdgcn_wmma_f32_16x16x32_bf16(false, (A), false, (B),           \
                                            (short)0, (C), false, false)

// ------------------ kernel 2: tiled GEMM + fused loss epilogue -------------
// Block = 256 threads (8 wave32), output tile 128(M) x 64(N).
// Waves arranged 4x2; each wave owns a 32x32 tile = 2x2 WMMA accumulators.
__global__ void __launch_bounds__(256)
cl_gemm_loss(const unsigned short* __restrict__ Ehi,
             const unsigned short* __restrict__ Elo,
             const int* __restrict__ labels,
             float* __restrict__ partials) {
    __shared__ int   lrow[128];
    __shared__ int   lcol[64];
    __shared__ float red[256];

    const int tid  = threadIdx.x;
    const int lane = tid & 31;
    const int w    = tid >> 5;
    const int wm   = w >> 1;          // 0..3
    const int wn   = w & 1;           // 0..1
    const int blockRow0 = blockIdx.y * 128;
    const int blockCol0 = blockIdx.x * 64;
    const int rowBase = blockRow0 + wm * 32;
    const int colBase = blockCol0 + wn * 32;

    if (tid < 128)                lrow[tid]       = labels[blockRow0 + tid];
    else if (tid < 192)           lcol[tid - 128] = labels[blockCol0 + (tid - 128)];
    __syncthreads();

    v8f acc00 = {}, acc01 = {}, acc10 = {}, acc11 = {};

    for (int kb = 0; kb < D_EMB; kb += 32) {
        v16bf ah0 = load_a_frag(Ehi, rowBase,      kb, lane);
        v16bf ah1 = load_a_frag(Ehi, rowBase + 16, kb, lane);
        v16bf al0 = load_a_frag(Elo, rowBase,      kb, lane);
        v16bf al1 = load_a_frag(Elo, rowBase + 16, kb, lane);
        v16bf bh0 = load_b_frag(Ehi, colBase,      kb, lane);
        v16bf bh1 = load_b_frag(Ehi, colBase + 16, kb, lane);
        v16bf bl0 = load_b_frag(Elo, colBase,      kb, lane);
        v16bf bl1 = load_b_frag(Elo, colBase + 16, kb, lane);

        acc00 = WMMA_BF16(ah0, bh0, acc00);
        acc00 = WMMA_BF16(ah0, bl0, acc00);
        acc00 = WMMA_BF16(al0, bh0, acc00);

        acc01 = WMMA_BF16(ah0, bh1, acc01);
        acc01 = WMMA_BF16(ah0, bl1, acc01);
        acc01 = WMMA_BF16(al0, bh1, acc01);

        acc10 = WMMA_BF16(ah1, bh0, acc10);
        acc10 = WMMA_BF16(ah1, bl0, acc10);
        acc10 = WMMA_BF16(al1, bh0, acc10);

        acc11 = WMMA_BF16(ah1, bh1, acc11);
        acc11 = WMMA_BF16(ah1, bl1, acc11);
        acc11 = WMMA_BF16(al1, bh1, acc11);
    }

    // Epilogue: C/D 16x16 layout — VGPR r: lanes 0-15 -> M=r, lanes 16-31 -> M=r+8;
    // N = lane & 15.
    const int mOff = (lane >> 4) << 3;
    const int ncol = lane & 15;
    float sum = 0.0f;

    #pragma unroll
    for (int mi = 0; mi < 2; ++mi) {
        #pragma unroll
        for (int ni = 0; ni < 2; ++ni) {
            v8f a = (mi == 0) ? ((ni == 0) ? acc00 : acc01)
                              : ((ni == 0) ? acc10 : acc11);
            #pragma unroll
            for (int r = 0; r < 8; ++r) {
                const int li = wm * 32 + mi * 16 + mOff + r;   // local row in [0,128)
                const int lj = wn * 32 + ni * 16 + ncol;       // local col in [0,64)
                const int gi = blockRow0 + li;
                const int gj = blockCol0 + lj;
                const float s = a[r];
                if (gi != gj) {
                    if (lrow[li] == lcol[lj]) {
                        const float d = 1.0f - s;
                        sum += d * d;
                    } else {
                        float d = s - MARGIN;
                        d = d > 0.0f ? d : 0.0f;
                        sum += d * d;
                    }
                }
            }
        }
    }

    red[tid] = sum;
    __syncthreads();
    #pragma unroll
    for (int s = 128; s > 0; s >>= 1) {
        if (tid < s) red[tid] += red[tid + s];
        __syncthreads();
    }
    if (tid == 0) partials[blockIdx.y * gridDim.x + blockIdx.x] = red[0];
}

// ------------------------ kernel 3: final reduction ------------------------
__global__ void cl_finalize(const float* __restrict__ partials,
                            float* __restrict__ out, int count) {
    __shared__ float red[256];
    const int tid = threadIdx.x;
    float s = 0.0f;
    for (int i = tid; i < count; i += 256) s += partials[i];
    red[tid] = s;
    __syncthreads();
    #pragma unroll
    for (int k = 128; k > 0; k >>= 1) {
        if (tid < k) red[tid] += red[tid + k];
        __syncthreads();
    }
    if (tid == 0)
        out[0] = red[0] * (1.0f / ((float)N_EMB * (float)(N_EMB - 1)));
}

// ------------------------------ launcher -----------------------------------
extern "C" void kernel_launch(void* const* d_in, const int* in_sizes, int n_in,
                              void* d_out, int out_size, void* d_ws, size_t ws_size,
                              hipStream_t stream) {
    (void)in_sizes; (void)n_in; (void)out_size; (void)ws_size;

    const float* x      = (const float*)d_in[0];
    const int*   labels = (const int*)d_in[1];
    float*       out    = (float*)d_out;

    unsigned short* Ehi = (unsigned short*)d_ws;
    unsigned short* Elo = Ehi + (size_t)N_EMB * D_EMB;
    float* partials     = (float*)(Elo + (size_t)N_EMB * D_EMB);

    const int gx = N_EMB / 64;    // 128
    const int gy = N_EMB / 128;   // 64

    cl_normalize<<<N_EMB, 256, 0, stream>>>(x, Ehi, Elo);
    cl_gemm_loss<<<dim3(gx, gy), 256, 0, stream>>>(Ehi, Elo, labels, partials);
    cl_finalize<<<1, 256, 0, stream>>>(partials, out, gx * gy);
}
